// CustomGRUCell_6030134084275
// MI455X (gfx1250) — compile-verified, matrix operand order checked by reading, not run on hardware
//
#include <hip/hip_runtime.h>
#include <hip/hip_bf16.h>

typedef __attribute__((ext_vector_type(16))) __bf16 v16bf;
typedef __attribute__((ext_vector_type(8)))  __bf16 v8bf;
typedef __attribute__((ext_vector_type(8)))  float  v8f;
typedef __attribute__((ext_vector_type(4)))  unsigned short us4;

#define M_TILE  64
#define THREADS 128
#define XPITCH  264      // bf16 elems per LDS row (256 + 8 pad -> 528B, 16B aligned)
#define ZPITCH  268      // f32 elems per LDS row
#define NT      16       // 256 / 16 output-column tiles
#define KT      8        // 256 / 32 K steps
#define W_ELEMS (256*256)

// ---- LDS layout (dynamic shared) ----
#define OFF_X   0
#define OFF_H   (M_TILE*XPITCH*2)                 // 33792
#define OFF_RH  (2*M_TILE*XPITCH*2)               // 67584
#define OFF_C   (3*M_TILE*XPITCH*2)               // 101376
#define OFF_Z   (4*M_TILE*XPITCH*2)               // 135168
#define OFF_CP  (OFF_Z + M_TILE*ZPITCH*4)         // 203776
#define SMEM_BYTES (OFF_CP + M_TILE*ZPITCH*4)     // 272384

__device__ __forceinline__ unsigned short f2bf(float f) {
    unsigned u = __float_as_uint(f);
    u += 0x7FFFu + ((u >> 16) & 1u);              // round-to-nearest-even
    return (unsigned short)(u >> 16);
}

__device__ __forceinline__ float sigm(float x) {
    return 1.0f / (1.0f + __expf(-x));
}

// Pre-swizzle one weight element into the WMMA 16-bit B-operand fragment layout:
// fragment(nt,kt) = 512 bf16: plane0 = lanes' K {0..7 | 8..15}, plane1 = K {16..23 | 24..31}
__global__ void prep_weights(const float* __restrict__ w0, const float* __restrict__ w1,
                             const float* __restrict__ w2, const float* __restrict__ w3,
                             const float* __restrict__ w4, const float* __restrict__ w5,
                             const float* __restrict__ w6, const float* __restrict__ w7,
                             unsigned short* __restrict__ wbf) {
    const float* Ws[8] = {w0, w1, w2, w3, w4, w5, w6, w7};
    int o = blockIdx.x * blockDim.x + threadIdx.x;      // [0, 8*65536)
    int mi    = o >> 16;
    int r     = o & 65535;
    int frag  = r >> 9;                                  // 128 fragments
    int nt    = frag >> 3;
    int kt    = frag & 7;
    int q     = r & 511;
    int plane = q >> 8;
    int L     = (q >> 3) & 31;
    int j     = q & 7;
    int n  = nt * 16 + (L & 15);
    int kb = kt * 32 + ((L < 16) ? (plane ? 16 : 0) : (plane ? 24 : 8));
    wbf[o] = f2bf(Ws[mi][n * 256 + kb + j]);
}

// Combined biases: bz = Wz_b+Uz_b, br = Wr_b+Ur_b, bc = Wh_b+Uh_b+bias_h, bg, bl
__global__ void prep_bias(const float* __restrict__ wzb, const float* __restrict__ uzb,
                          const float* __restrict__ wrb, const float* __restrict__ urb,
                          const float* __restrict__ whb, const float* __restrict__ uhb,
                          const float* __restrict__ gb,  const float* __restrict__ lb,
                          const float* __restrict__ bh,  float* __restrict__ out) {
    int n = threadIdx.x;
    out[n]        = wzb[n] + uzb[n];
    out[256 + n]  = wrb[n] + urb[n];
    out[512 + n]  = whb[n] + uhb[n] + bh[n];
    out[768 + n]  = gb[n];
    out[1024 + n] = lb[n];
}

__global__ void __launch_bounds__(THREADS)
gru_fused(const float* __restrict__ x, const float* __restrict__ h,
          const __bf16* __restrict__ wbf, const float* __restrict__ bias,
          float* __restrict__ out) {
    extern __shared__ char smem[];
    __bf16* xs  = (__bf16*)(smem + OFF_X);
    __bf16* hs  = (__bf16*)(smem + OFF_H);
    __bf16* rhs = (__bf16*)(smem + OFF_RH);
    __bf16* cs  = (__bf16*)(smem + OFF_C);
    float*  zs  = (float*)(smem + OFF_Z);
    float*  cp  = (float*)(smem + OFF_CP);

    const int tid  = threadIdx.x;
    const int lane = tid & 31;
    const int wave = tid >> 5;
    const int hf   = lane >> 4;      // half-wave
    const int mr   = lane & 15;
    const long row0 = (long)blockIdx.x * M_TILE;
    const int  lr0  = wave * 16;     // this wave's 16 LDS rows

    // ---- cooperative tile load: global f32 -> LDS bf16 ----
    const float4* xg = (const float4*)(x + row0 * 256);
    const float4* hg = (const float4*)(h + row0 * 256);
    for (int i = tid; i < M_TILE * 64; i += THREADS) {
        int r = i >> 6, c4 = i & 63;
        float4 vx = xg[i];
        float4 vh = hg[i];
        us4 px = { f2bf(vx.x), f2bf(vx.y), f2bf(vx.z), f2bf(vx.w) };
        us4 ph = { f2bf(vh.x), f2bf(vh.y), f2bf(vh.z), f2bf(vh.w) };
        *(us4*)(xs + r * XPITCH + c4 * 4) = px;
        *(us4*)(hs + r * XPITCH + c4 * 4) = ph;
    }
    __syncthreads();

    union AF { v16bf v; v8bf p[2]; };

    // A fragment: lane mr holds row lr0+mr; half 0: K {0..7,16..23}, half 1: K {8..15,24..31}
    #define LOAD_A(buf, kt_, frag) do {                                            \
        const __bf16* bp_ = (buf) + (lr0 + mr) * XPITCH + (kt_) * 32 + hf * 8;     \
        (frag).p[0] = *(const v8bf*)(bp_);                                         \
        (frag).p[1] = *(const v8bf*)(bp_ + 16);                                    \
    } while (0)

    // B fragment from pre-swizzled global weights (coalesced 16B per lane, L0/L2 resident)
    #define LOAD_B(mi_, nt_, kt_, frag) do {                                       \
        const __bf16* bp_ = wbf + (mi_) * W_ELEMS + (((nt_) * 8 + (kt_)) * 512)    \
                            + lane * 8;                                            \
        (frag).p[0] = *(const v8bf*)(bp_);                                         \
        (frag).p[1] = *(const v8bf*)(bp_ + 256);                                   \
    } while (0)

    #define WMMA(A_, B_, C_) __builtin_amdgcn_wmma_f32_16x16x32_bf16(              \
        false, (A_).v, false, (B_).v, (short)0, (C_), false, false)

    const float* bz = bias;
    const float* br = bias + 256;
    const float* bc = bias + 512;
    const float* bg = bias + 768;
    const float* bl = bias + 1024;

    // ================= Stage 1: z, r, x@Wh^T (one pass over x & h) =================
    for (int nt = 0; nt < NT; ++nt) {
        v8f az = {}; v8f ar = {}; v8f ah = {};
        #pragma unroll
        for (int kt = 0; kt < KT; ++kt) {
            AF ax, axh, b;
            LOAD_A(xs, kt, ax);
            LOAD_A(hs, kt, axh);
            LOAD_B(0, nt, kt, b); az = WMMA(ax,  b, az);   // x @ Wz^T
            LOAD_B(1, nt, kt, b); az = WMMA(axh, b, az);   // h @ Uz^T
            LOAD_B(2, nt, kt, b); ar = WMMA(ax,  b, ar);   // x @ Wr^T
            LOAD_B(3, nt, kt, b); ar = WMMA(axh, b, ar);   // h @ Ur^T
            LOAD_B(4, nt, kt, b); ah = WMMA(ax,  b, ah);   // x @ Wh^T
        }
        #pragma unroll
        for (int e = 0; e < 8; ++e) {
            int M  = e + hf * 8;
            int lr = lr0 + M;
            int n  = nt * 16 + mr;
            float zv = sigm(az[e] + bz[n]);
            float rv = sigm(ar[e] + br[n]);
            float hv = h[(row0 + lr) * 256 + n];
            zs[lr * ZPITCH + n] = zv;
            cp[lr * ZPITCH + n] = ah[e];                         // cand partial (f32)
            *(unsigned short*)(rhs + lr * XPITCH + n) = f2bf(rv * hv);
        }
    }
    // Wave-private rows only -> no barrier needed between stages.

    // ================= Stage 2: cand_in = partial + (r*h) @ Uh^T + bc =============
    for (int nt = 0; nt < NT; ++nt) {
        v8f ac = {};
        #pragma unroll
        for (int kt = 0; kt < KT; ++kt) {
            AF a, b;
            LOAD_A(rhs, kt, a);
            LOAD_B(5, nt, kt, b);
            ac = WMMA(a, b, ac);
        }
        #pragma unroll
        for (int e = 0; e < 8; ++e) {
            int M  = e + hf * 8;
            int lr = lr0 + M;
            int n  = nt * 16 + mr;
            float cv = ac[e] + cp[lr * ZPITCH + n] + bc[n];
            *(unsigned short*)(cs + lr * XPITCH + n) = f2bf(cv);
        }
    }

    // ================= Stage 3: gate/linear GEMMs + final blend ===================
    for (int nt = 0; nt < NT; ++nt) {
        v8f ag = {}; v8f al = {};
        #pragma unroll
        for (int kt = 0; kt < KT; ++kt) {
            AF a, b;
            LOAD_A(cs, kt, a);
            LOAD_B(6, nt, kt, b); ag = WMMA(a, b, ag);     // cand_in @ gate^T
            LOAD_B(7, nt, kt, b); al = WMMA(a, b, al);     // cand_in @ linear^T
        }
        #pragma unroll
        for (int e = 0; e < 8; ++e) {
            int M  = e + hf * 8;
            int lr = lr0 + M;
            int n  = nt * 16 + mr;
            float g    = sigm(ag[e] + bg[n]);
            float cand = (al[e] + bl[n]) * g;
            float zv   = zs[lr * ZPITCH + n];
            float hv   = h[(row0 + lr) * 256 + n];
            out[(row0 + lr) * 256 + n] = zv * cand + (1.0f - zv) * hv;
        }
    }

    #undef LOAD_A
    #undef LOAD_B
    #undef WMMA
}

extern "C" void kernel_launch(void* const* d_in, const int* in_sizes, int n_in,
                              void* d_out, int out_size, void* d_ws, size_t ws_size,
                              hipStream_t stream) {
    (void)n_in; (void)out_size; (void)ws_size;
    const float* x = (const float*)d_in[0];
    const float* h = (const float*)d_in[1];
    // dict order: x,h, then (w,b) pairs for W_z,U_z,W_r,U_r,W_h,U_h,lin_gate,lin_linear, bias_h
    unsigned short* wbf  = (unsigned short*)d_ws;                 // 8 * 65536 bf16 = 1 MB
    float*          bias = (float*)((char*)d_ws + 8 * W_ELEMS * 2);

    prep_weights<<<(8 * W_ELEMS) / 256, 256, 0, stream>>>(
        (const float*)d_in[2],  (const float*)d_in[4],  (const float*)d_in[6],
        (const float*)d_in[8],  (const float*)d_in[10], (const float*)d_in[12],
        (const float*)d_in[14], (const float*)d_in[16], wbf);

    prep_bias<<<1, 256, 0, stream>>>(
        (const float*)d_in[3],  (const float*)d_in[5],  (const float*)d_in[7],
        (const float*)d_in[9],  (const float*)d_in[11], (const float*)d_in[13],
        (const float*)d_in[15], (const float*)d_in[17], (const float*)d_in[18], bias);

    int rows = in_sizes[0] / 256;                 // 200000 (multiple of 64)
    int grid = (rows + M_TILE - 1) / M_TILE;      // 3125
    gru_fused<<<grid, THREADS, SMEM_BYTES, stream>>>(
        x, h, (const __bf16*)wbf, bias, (float*)d_out);
}
